// ComplEx_39565238731231
// MI455X (gfx1250) — matrix operand order repaired and checked by reading, not compiled, not run on hardware
//
#include <hip/hip_runtime.h>

// ComplEx scoring on MI455X (gfx1250).
//
// reference:  combined = bn0(e1)*(rel+rel) + bn1(e1)*(rel-rel)  ==  2*bn0(e1)*rel
//             out      = sigmoid(combined @ all_ent_emb^T)      [1024 x 100000]
//
// Strategy: split-bf16 (hi+lo) emulated-fp32 GEMM using v_wmma_f32_16x16x32_bf16:
//   a*b ~= a_hi*b_hi + a_hi*b_lo + a_lo*b_hi   (f32 accumulate, ~1e-5 rel. error)
// Grid ordered M-fastest so each E n-tile is shared across all 8 M-tiles via L2.
// All GEMM addressing uses uniform SGPR base + 32-bit BYTE lane offset so every
// load/store lowers to saddr-form global_load_b128 / global_store_b32 with
// immediate offsets (no in-loop 64-bit VALU address math).

#define B_DIM  1024
#define D_DIM  512
#define N_ENT  100000
#define K_DIM  512
#define WG_M   128
#define WG_N   128
#define BN_EPS 1e-5f

typedef __attribute__((ext_vector_type(16))) __bf16 v16bf;
typedef __attribute__((ext_vector_type(8)))  __bf16 v8bf;
typedef __attribute__((ext_vector_type(4)))  __bf16 v4bf;
typedef __attribute__((ext_vector_type(8)))  float  v8f;
typedef __attribute__((ext_vector_type(4)))  float  v4f;

union FragA { v16bf v; v8bf h[2]; };

// ---------------------------------------------------------------------------
// Kernel 1: per-column BatchNorm stats over e1_emb (axis 0), folded into
// scale/shift:  bn0(x)[b,d] = x[b,d]*scale0[d] + shift0[d]
// ---------------------------------------------------------------------------
__global__ void bn_stats_kernel(const float* __restrict__ e1,
                                const float* __restrict__ gamma0,
                                const float* __restrict__ beta0,
                                float* __restrict__ scale0,
                                float* __restrict__ shift0) {
    int d = blockIdx.x * blockDim.x + threadIdx.x;
    if (d >= D_DIM) return;
    float s = 0.f, sq = 0.f;
    for (int b = 0; b < B_DIM; ++b) {
        float x = e1[(unsigned)(b * D_DIM + d)];
        s += x;
        sq += x * x;
    }
    float mean = s * (1.f / B_DIM);
    float var  = sq * (1.f / B_DIM) - mean * mean;
    float sc   = gamma0[d] * rsqrtf(var + BN_EPS);
    scale0[d]  = sc;
    shift0[d]  = beta0[d] - mean * sc;
}

// ---------------------------------------------------------------------------
// Kernel 2: A = 2 * rel * bn0(e1)  ->  split into bf16 hi + bf16 lo
// ---------------------------------------------------------------------------
__global__ void prep_a_kernel(const float* __restrict__ e1,
                              const float* __restrict__ rel,
                              const float* __restrict__ scale0,
                              const float* __restrict__ shift0,
                              __bf16* __restrict__ Ahi,
                              __bf16* __restrict__ Alo) {
    unsigned idx = blockIdx.x * blockDim.x + threadIdx.x;   // quad index
    v4f x = ((const v4f*)e1)[idx];
    v4f r = ((const v4f*)rel)[idx];
    int d4 = (idx & (D_DIM / 4 - 1)) * 4;
    v4bf hi, lo;
#pragma unroll
    for (int j = 0; j < 4; ++j) {
        float c = 2.f * r[j] * (x[j] * scale0[d4 + j] + shift0[d4 + j]);
        __bf16 h = (__bf16)c;
        hi[j] = h;
        lo[j] = (__bf16)(c - (float)h);
    }
    ((v4bf*)Ahi)[idx] = hi;
    ((v4bf*)Alo)[idx] = lo;
}

// ---------------------------------------------------------------------------
// Kernel 3: all_ent_emb f32 -> bf16 hi + bf16 lo
// ---------------------------------------------------------------------------
__global__ void prep_e_kernel(const float* __restrict__ E,
                              __bf16* __restrict__ Ehi,
                              __bf16* __restrict__ Elo) {
    unsigned idx = blockIdx.x * blockDim.x + threadIdx.x;   // quad index
    v4f x = ((const v4f*)E)[idx];
    v4bf hi, lo;
#pragma unroll
    for (int j = 0; j < 4; ++j) {
        __bf16 h = (__bf16)x[j];
        hi[j] = h;
        lo[j] = (__bf16)(x[j] - (float)h);
    }
    ((v4bf*)Ehi)[idx] = hi;
    ((v4bf*)Elo)[idx] = lo;
}

// ---------------------------------------------------------------------------
// Kernel 4: WMMA GEMM  out = sigmoid(A @ E^T), M=1024 K=512 N=100000
// Block = 8 waves (4 M x 2 N), wave tile = 32x64 (2x4 subtiles of 16x16).
// 3 WMMAs per subtile per K-chunk (hi*hi, hi*lo, lo*hi).
// ---------------------------------------------------------------------------
__global__ void __launch_bounds__(256)
gemm_wmma_kernel(const __bf16* __restrict__ Ahi, const __bf16* __restrict__ Alo,
                 const __bf16* __restrict__ Ehi, const __bf16* __restrict__ Elo,
                 float* __restrict__ out) {
    const int m_tiles = B_DIM / WG_M;          // 8; M is the fast grid dim
    int m_tile = blockIdx.x % m_tiles;
    int n_tile = blockIdx.x / m_tiles;

    int wave = threadIdx.x >> 5;
    int lane = threadIdx.x & 31;
    int hl   = lane >> 4;                      // half-wave: 0 or 1
    int l16  = lane & 15;

    int wave_m = wave >> 1;                    // 0..3
    int wave_n = wave & 1;                     // 0..1
    int m0 = m_tile * WG_M + wave_m * 32;
    int n0 = n_tile * WG_N + wave_n * 64;

    v8f zero = {0.f, 0.f, 0.f, 0.f, 0.f, 0.f, 0.f, 0.f};
    v8f acc[2][4];
#pragma unroll
    for (int i = 0; i < 2; ++i)
#pragma unroll
        for (int j = 0; j < 4; ++j) acc[i][j] = zero;

    const char* AhiB = (const char*)Ahi;
    const char* AloB = (const char*)Alo;
    const char* EhiB = (const char*)Ehi;
    const char* EloB = (const char*)Elo;

    // Per-lane 32-bit BYTE offsets (all buffers < 4 GiB): uniform SGPR base
    // + zext(u32) VGPR byte offset + immediate -> saddr-form global_load_b128.
    // A fragment (16-bit A 16x32): lane row = m0+l16; lanes 0-15 take
    // K 0..15, lanes 16-31 take K 16..31.
    unsigned aByte = (unsigned)((m0 + l16) * (K_DIM * 2) + hl * 32);
    // B fragment (16-bit B 32x16): lane column = l16 (E row); lanes 0-15 hold
    // K 0..15, lanes 16-31 hold K 16..31 -> one contiguous 32B run per lane.
    unsigned bByte[4];
#pragma unroll
    for (int ns = 0; ns < 4; ++ns) {
        int n = n0 + ns * 16 + l16;
        n = n < N_ENT ? n : N_ENT - 1;         // clamp (stores are guarded)
        bByte[ns] = (unsigned)(n * (K_DIM * 2) + hl * 32);
    }

#pragma unroll
    for (int kb = 0; kb < K_DIM; kb += 32) {
        FragA ah[2], al[2];
#pragma unroll
        for (int ms = 0; ms < 2; ++ms) {
            int c0 = ms * 16 * K_DIM * 2 + kb * 2;   // immediate part (bytes)
            ah[ms].h[0] = *(const v8bf*)(AhiB + aByte + c0);
            ah[ms].h[1] = *(const v8bf*)(AhiB + aByte + c0 + 16);
            al[ms].h[0] = *(const v8bf*)(AloB + aByte + c0);
            al[ms].h[1] = *(const v8bf*)(AloB + aByte + c0 + 16);
        }
#pragma unroll
        for (int ns = 0; ns < 4; ++ns) {
            v16bf bh = *(const v16bf*)(EhiB + bByte[ns] + kb * 2);
            v16bf bl = *(const v16bf*)(EloB + bByte[ns] + kb * 2);
#pragma unroll
            for (int ms = 0; ms < 2; ++ms) {
                acc[ms][ns] = __builtin_amdgcn_wmma_f32_16x16x32_bf16(
                    false, ah[ms].v, false, bh, (short)0, acc[ms][ns], false, false);
                acc[ms][ns] = __builtin_amdgcn_wmma_f32_16x16x32_bf16(
                    false, ah[ms].v, false, bl, (short)0, acc[ms][ns], false, false);
                acc[ms][ns] = __builtin_amdgcn_wmma_f32_16x16x32_bf16(
                    false, al[ms].v, false, bh, (short)0, acc[ms][ns], false, false);
            }
        }
    }

    // C/D layout: VGPR r, lanes 0-15 -> M=r, N=lane; lanes 16-31 -> M=r+8.
    char* outB = (char*)out;
#pragma unroll
    for (int ms = 0; ms < 2; ++ms) {
#pragma unroll
        for (int ns = 0; ns < 4; ++ns) {
            int n = n0 + ns * 16 + l16;
            if (n < N_ENT) {
                unsigned oByte =
                    (unsigned)(((m0 + ms * 16 + hl * 8) * N_ENT + n) * 4);
#pragma unroll
                for (int r = 0; r < 8; ++r) {
                    float x = acc[ms][ns][r];
                    *(float*)(outB + oByte + r * (N_ENT * 4)) =
                        1.f / (1.f + __expf(-x));
                }
            }
        }
    }
}

// ---------------------------------------------------------------------------
extern "C" void kernel_launch(void* const* d_in, const int* in_sizes, int n_in,
                              void* d_out, int out_size, void* d_ws, size_t ws_size,
                              hipStream_t stream) {
    const float* e1     = (const float*)d_in[0];
    const float* rel    = (const float*)d_in[1];
    const float* ent    = (const float*)d_in[2];
    const float* gamma0 = (const float*)d_in[3];
    const float* beta0  = (const float*)d_in[4];
    // d_in[5]/d_in[6] (gamma1/beta1) feed a term multiplied by rel-rel == 0.

    char* ws = (char*)d_ws;
    float*  scale0 = (float*)(ws + 0);
    float*  shift0 = (float*)(ws + 2048);
    __bf16* Ahi    = (__bf16*)(ws + 4096);
    __bf16* Alo    = (__bf16*)(ws + 4096 + (size_t)B_DIM * D_DIM * 2);
    __bf16* Ehi    = (__bf16*)(ws + 4096 + (size_t)B_DIM * D_DIM * 4);
    __bf16* Elo    = Ehi + (size_t)N_ENT * D_DIM;

    bn_stats_kernel<<<(D_DIM + 255) / 256, 256, 0, stream>>>(e1, gamma0, beta0,
                                                             scale0, shift0);

    int a_quads = B_DIM * D_DIM / 4;                       // 131072
    prep_a_kernel<<<a_quads / 256, 256, 0, stream>>>(e1, rel, scale0, shift0,
                                                     Ahi, Alo);

    size_t e_quads = (size_t)N_ENT * D_DIM / 4;            // 12,800,000
    prep_e_kernel<<<(unsigned)(e_quads / 256), 256, 0, stream>>>(ent, Ehi, Elo);

    int n_tiles = (N_ENT + WG_N - 1) / WG_N;               // 782
    gemm_wmma_kernel<<<(B_DIM / WG_M) * n_tiles, 256, 0, stream>>>(
        Ahi, Alo, Ehi, Elo, (float*)d_out);
}
